// EnergyBasedMultiHeadAttention_86157043957881
// MI455X (gfx1250) — compile-verified, matrix operand order checked by reading, not run on hardware
//
#include <hip/hip_runtime.h>
#include <hip/hip_bf16.h>

#define DEV __device__ __forceinline__

typedef __bf16 bf16;
typedef __attribute__((ext_vector_type(16))) __bf16 v16bf;
typedef __attribute__((ext_vector_type(8)))  float   v8f;

namespace {
constexpr int   Bn  = 2;
constexpr int   Ln  = 2048;
constexpr int   Dn  = 1024;
constexpr int   Hn  = 16;
constexpr int   DKn = 64;
constexpr int   Mn  = Bn * Ln;          // 4096 token rows
constexpr int   LDSW = Ln + 4;          // padded LDS row stride (floats)
constexpr float kQKScale  = 0.125f;     // 1/sqrt(64)
constexpr float kEnergy   = 0.1f;
constexpr float kEps      = 1e-5f;
}

// ---------------------------------------------------------------------------
// WMMA tile loaders (wave32).  Layouts per CDNA5 ISA §7.12.2:
//   16-bit A 16x32:  lanes 0-15 -> M=lane, K in {0..7,16..23};
//                    lanes 16-31 -> M=lane-16, K in {8..15,24..31}
//   B 32x16 mirrors A (column n of B contiguous in K for NT operands)
//   f32 C/D 16x16:   VGPR r -> M = r + 8*(lane>>4), N = lane&15
// ---------------------------------------------------------------------------
union ABf { v16bf v; uint4 q[2]; bf16 e[16]; };

// A[16x32] (row-major, ld elems) or NT-B (W [N x K] row-major): contiguous in
// K per lane -> two b128 loads.
DEV v16bf load_tile_nt_bf16(const bf16* __restrict__ tile, int ld) {
  const int lane = threadIdx.x & 31;
  const int row  = lane & 15;
  const int kb   = (lane >> 4) << 3;   // 0 or 8
  const bf16* p  = tile + (long)row * ld + kb;
  ABf r;
  r.q[0] = *reinterpret_cast<const uint4*>(p);
  r.q[1] = *reinterpret_cast<const uint4*>(p + 16);
  return r.v;
}

// A[16x32] from an f32 buffer (LDS-resident attn rows), cvt to bf16 in-reg.
DEV v16bf load_tile_f32cvt(const float* tile, int ld) {
  const int lane = threadIdx.x & 31;
  const int row  = lane & 15;
  const int kb   = (lane >> 4) << 3;
  const float* p = tile + (long)row * ld + kb;
  ABf r;
#pragma unroll
  for (int j = 0; j < 8; ++j) {
    r.e[j]     = (bf16)p[j];
    r.e[8 + j] = (bf16)p[16 + j];
  }
  return r.v;
}

DEV v8f wmma_bf16(v16bf a, v16bf b, v8f c) {
  return __builtin_amdgcn_wmma_f32_16x16x32_bf16(
      /*neg_a=*/false, a, /*neg_b=*/false, b,
      /*c_mod=*/(short)0, c, /*reuse_a=*/false, /*reuse_b=*/false);
}

DEV v8f zero8() { v8f z = {0.f,0.f,0.f,0.f,0.f,0.f,0.f,0.f}; return z; }

// ---------------------------------------------------------------------------
// Reductions
// ---------------------------------------------------------------------------
DEV float wave_sum(float v) {
#pragma unroll
  for (int o = 16; o > 0; o >>= 1) v += __shfl_down(v, o, 32);
  return v;
}
DEV float wave_max(float v) {
#pragma unroll
  for (int o = 16; o > 0; o >>= 1) v = fmaxf(v, __shfl_down(v, o, 32));
  return v;
}
DEV float block_sum(float v, float* sm) {
  const int w = threadIdx.x >> 5, l = threadIdx.x & 31;
  v = wave_sum(v);
  if (l == 0) sm[w] = v;
  __syncthreads();
  if (w == 0) {
    float t = (l < 8) ? sm[l] : 0.f;
    t = wave_sum(t);
    if (l == 0) sm[0] = t;
  }
  __syncthreads();
  v = sm[0];
  __syncthreads();
  return v;
}

// ---------------------------------------------------------------------------
// Kernels
// ---------------------------------------------------------------------------

// f32 -> bf16 bulk convert, 4 elems/thread
__global__ void __launch_bounds__(256)
k_cvt_bf16(const float* __restrict__ s, bf16* __restrict__ d, int n4) {
  int i = blockIdx.x * 256 + threadIdx.x;
  if (i >= n4) return;
  float4 f = reinterpret_cast<const float4*>(s)[i];
  union { bf16 h[4]; uint2 u; } t;
  t.h[0] = (bf16)f.x; t.h[1] = (bf16)f.y; t.h[2] = (bf16)f.z; t.h[3] = (bf16)f.w;
  reinterpret_cast<uint2*>(d)[i] = t.u;
}

// Generic batched NT GEMM, register-blocked 1x4 tiles per wave:
//   C[m, n0..n0+63] = sum_k A[m,k] * Bw[n,k] (+bias[n]).
// All 10 b128 loads of a k-step are issued before the 4 WMMAs so the
// scheduler can use progressive s_wait_loadcnt values.
__global__ void __launch_bounds__(256)
k_gemm_nt(const bf16* __restrict__ A, long sAb, int lda,
          const bf16* __restrict__ Bw, long sBb, int ldb,
          float* __restrict__ Cf, bf16* __restrict__ Cbf, long sCb, int ldc,
          const float* __restrict__ bias, int K) {
  const int wave = threadIdx.x >> 5;
  const int tn0  = (blockIdx.x * 8 + wave) * 4;   // first of 4 N tiles
  const int tm   = blockIdx.y;
  const int bz   = blockIdx.z;
  const bf16* a0 = A  + (long)bz * sAb + (long)tm * 16 * lda;
  const bf16* b0 = Bw + (long)bz * sBb + (long)tn0 * 16 * ldb;
  v8f acc[4] = {zero8(), zero8(), zero8(), zero8()};
  for (int k = 0; k < K; k += 32) {
    v16bf a  = load_tile_nt_bf16(a0 + k, lda);
    v16bf t0 = load_tile_nt_bf16(b0 + k, ldb);
    v16bf t1 = load_tile_nt_bf16(b0 + (long)16 * ldb + k, ldb);
    v16bf t2 = load_tile_nt_bf16(b0 + (long)32 * ldb + k, ldb);
    v16bf t3 = load_tile_nt_bf16(b0 + (long)48 * ldb + k, ldb);
    acc[0] = wmma_bf16(a, t0, acc[0]);
    acc[1] = wmma_bf16(a, t1, acc[1]);
    acc[2] = wmma_bf16(a, t2, acc[2]);
    acc[3] = wmma_bf16(a, t3, acc[3]);
  }
  const int lane = threadIdx.x & 31;
  const int col  = lane & 15;
  const int rb   = (lane >> 4) << 3;
#pragma unroll
  for (int j = 0; j < 4; ++j) {
    const float bv = bias ? bias[(tn0 + j) * 16 + col] : 0.f;
    const long cbase =
        (long)bz * sCb + ((long)tm * 16) * ldc + (tn0 + j) * 16;
#pragma unroll
    for (int r = 0; r < 8; ++r) {
      float v = acc[j][r] + bv;
      long idx = cbase + (long)(rb + r) * ldc + col;
      if (Cf)  Cf[idx]  = v;
      if (Cbf) Cbf[idx] = (bf16)v;
    }
  }
}

// transpose V per (b,h): vb[B,L,D] slice [L x dk] -> vt[bh][dk x L]
__global__ void __launch_bounds__(256)
k_vtrans(const bf16* __restrict__ vb, bf16* __restrict__ vt) {
  __shared__ bf16 t[64][66];
  const int bh = blockIdx.y, bb = bh >> 4, h = bh & 15;
  const int l0 = blockIdx.x * 64;
  {
    const int d  = threadIdx.x & 63;     // coalesced in d
    const int lr = threadIdx.x >> 6;     // 0..3
#pragma unroll
    for (int i = 0; i < 16; ++i) {
      int l = lr + i * 4;
      t[l][d] = vb[((long)(bb * Ln) + l0 + l) * Dn + h * DKn + d];
    }
  }
  __syncthreads();
  {
    const int l  = threadIdx.x & 63;     // coalesced in l
    const int dr = threadIdx.x >> 6;
#pragma unroll
    for (int i = 0; i < 16; ++i) {
      int d = dr + i * 4;
      vt[((long)bh * DKn + d) * Ln + l0 + l] = t[l][d];
    }
  }
}

// ---------------------------------------------------------------------------
// Fused attention kernel: scores -> LDS, softmax in LDS, attn written once to
// d_out, ctx computed from LDS-resident attn.  One block = 16 query rows of
// one (b,h).  Dynamic LDS: 16 x LDSW floats (~128 KB of the 320 KB WGP pool).
// ---------------------------------------------------------------------------
__global__ void __launch_bounds__(256)
k_attn_fused(const bf16* __restrict__ qb, const bf16* __restrict__ kb,
             const bf16* __restrict__ vt, const float* __restrict__ ebias,
             float* __restrict__ attn, bf16* __restrict__ ctx) {
  extern __shared__ float smem[];           // [16][LDSW]
  const int wave = threadIdx.x >> 5;
  const int lane = threadIdx.x & 31;
  const int tm   = blockIdx.x;              // 16-row tile of queries
  const int bh   = blockIdx.y;
  const int bb   = bh >> 4, h = bh & 15;
  const int col  = lane & 15;
  const int rb   = (lane >> 4) << 3;

  // ---- Phase A: scores (q k^T/sqrt(dk) + 0.1*ebias) into LDS ----
  const bf16* a0 = qb + ((long)(bb * Ln) + tm * 16) * Dn + h * DKn;
  for (int s = wave; s < Ln / 64; s += 8) {   // 32 strips of 64 columns
    const int tn0 = s * 4;
    const bf16* b0 = kb + ((long)(bb * Ln) + tn0 * 16) * Dn + h * DKn;
    v8f acc[4] = {zero8(), zero8(), zero8(), zero8()};
#pragma unroll
    for (int k = 0; k < DKn; k += 32) {
      v16bf a  = load_tile_nt_bf16(a0 + k, Dn);
      v16bf t0 = load_tile_nt_bf16(b0 + k, Dn);
      v16bf t1 = load_tile_nt_bf16(b0 + (long)16 * Dn + k, Dn);
      v16bf t2 = load_tile_nt_bf16(b0 + (long)32 * Dn + k, Dn);
      v16bf t3 = load_tile_nt_bf16(b0 + (long)48 * Dn + k, Dn);
      acc[0] = wmma_bf16(a, t0, acc[0]);
      acc[1] = wmma_bf16(a, t1, acc[1]);
      acc[2] = wmma_bf16(a, t2, acc[2]);
      acc[3] = wmma_bf16(a, t3, acc[3]);
    }
#pragma unroll
    for (int j = 0; j < 4; ++j) {
      const float* eb =
          ebias + ((long)bb * Ln + tm * 16) * Ln + (tn0 + j) * 16;
#pragma unroll
      for (int r = 0; r < 8; ++r)
        smem[(rb + r) * LDSW + (tn0 + j) * 16 + col] =
            kQKScale * acc[j][r] + kEnergy * eb[(long)(rb + r) * Ln + col];
    }
  }
  __syncthreads();

  // ---- Phase B: softmax per row (2 rows per wave), write attn once ----
#pragma unroll
  for (int rr = 0; rr < 2; ++rr) {
    const int r = wave * 2 + rr;
    float4* row = reinterpret_cast<float4*>(smem + r * LDSW);
    float m = -INFINITY;
#pragma unroll
    for (int i = 0; i < 16; ++i) {
      float4 u = row[lane + 32 * i];
      m = fmaxf(m, fmaxf(fmaxf(u.x, u.y), fmaxf(u.z, u.w)));
    }
    m = wave_max(m);
    float s = 0.f;
#pragma unroll
    for (int i = 0; i < 16; ++i) {
      float4 u = row[lane + 32 * i];
      u.x = __expf(u.x - m); u.y = __expf(u.y - m);
      u.z = __expf(u.z - m); u.w = __expf(u.w - m);
      s += (u.x + u.y) + (u.z + u.w);
      row[lane + 32 * i] = u;
    }
    s = wave_sum(s);
    const float inv = 1.f / s;
    float4* grow = reinterpret_cast<float4*>(
        attn + ((long)bh * Ln + tm * 16 + r) * Ln);
#pragma unroll
    for (int i = 0; i < 16; ++i) {
      float4 u = row[lane + 32 * i];
      u.x *= inv; u.y *= inv; u.z *= inv; u.w *= inv;
      row[lane + 32 * i]  = u;   // for ctx phase
      grow[lane + 32 * i] = u;   // attention-probs output
    }
  }
  __syncthreads();

  // ---- Phase C: ctx = attn(LDS) @ v ; one dk tile (16 cols) per wave ----
  if (wave < 4) {
    const int j = wave;
    const bf16* b0 = vt + (long)bh * DKn * Ln + (long)j * 16 * Ln;
    v8f acc = zero8();
    for (int k = 0; k < Ln; k += 32)
      acc = wmma_bf16(load_tile_f32cvt(smem + k, LDSW),
                      load_tile_nt_bf16(b0 + k, Ln), acc);
#pragma unroll
    for (int r = 0; r < 8; ++r)
      ctx[((long)(bb * Ln) + tm * 16 + rb + r) * Dn + h * DKn + j * 16 + col] =
          (bf16)acc[r];
  }
}

// layernorm over D=1024, one float4 per thread -> bf16 out (energy features)
__global__ void __launch_bounds__(256)
k_ln_bf16(const float* __restrict__ src, const float* __restrict__ g,
          const float* __restrict__ b, bf16* __restrict__ dst) {
  __shared__ float sm[8];
  const float4 v = reinterpret_cast<const float4*>(src + (long)blockIdx.x * Dn)
                       [threadIdx.x];
  const float mean = block_sum(v.x + v.y + v.z + v.w, sm) * (1.f / Dn);
  const float dx = v.x - mean, dy = v.y - mean, dz = v.z - mean, dw = v.w - mean;
  const float var =
      block_sum(dx * dx + dy * dy + dz * dz + dw * dw, sm) * (1.f / Dn);
  const float rs = rsqrtf(var + kEps);
  const float4 gv = reinterpret_cast<const float4*>(g)[threadIdx.x];
  const float4 bv = reinterpret_cast<const float4*>(b)[threadIdx.x];
  union { bf16 h[4]; uint2 u; } o;
  o.h[0] = (bf16)(dx * rs * gv.x + bv.x);
  o.h[1] = (bf16)(dy * rs * gv.y + bv.y);
  o.h[2] = (bf16)(dz * rs * gv.z + bv.z);
  o.h[3] = (bf16)(dw * rs * gv.w + bv.w);
  reinterpret_cast<uint2*>(dst + (long)blockIdx.x * Dn)[threadIdx.x] = o.u;
}

// final: y = LN(out0 + x) * g + b -> f32, one float4 per thread
__global__ void __launch_bounds__(256)
k_ln_residual(const float* __restrict__ o0, const float* __restrict__ x,
              const float* __restrict__ g, const float* __restrict__ b,
              float* __restrict__ y) {
  __shared__ float sm[8];
  const long rowo = (long)blockIdx.x * Dn;
  const float4 a = reinterpret_cast<const float4*>(o0 + rowo)[threadIdx.x];
  const float4 c = reinterpret_cast<const float4*>(x + rowo)[threadIdx.x];
  float4 v = {a.x + c.x, a.y + c.y, a.z + c.z, a.w + c.w};
  const float mean = block_sum(v.x + v.y + v.z + v.w, sm) * (1.f / Dn);
  const float dx = v.x - mean, dy = v.y - mean, dz = v.z - mean, dw = v.w - mean;
  const float var =
      block_sum(dx * dx + dy * dy + dz * dz + dw * dw, sm) * (1.f / Dn);
  const float rs = rsqrtf(var + kEps);
  const float4 gv = reinterpret_cast<const float4*>(g)[threadIdx.x];
  const float4 bv = reinterpret_cast<const float4*>(b)[threadIdx.x];
  float4 o = {dx * rs * gv.x + bv.x, dy * rs * gv.y + bv.y,
              dz * rs * gv.z + bv.z, dw * rs * gv.w + bv.w};
  reinterpret_cast<float4*>(y + rowo)[threadIdx.x] = o;
}

// ---------------------------------------------------------------------------
// Host launcher
// ---------------------------------------------------------------------------
extern "C" void kernel_launch(void* const* d_in, const int* in_sizes, int n_in,
                              void* d_out, int out_size, void* d_ws, size_t ws_size,
                              hipStream_t stream) {
  (void)in_sizes; (void)n_in; (void)out_size; (void)ws_size;
  const float* x    = (const float*)d_in[0];
  const float* wq   = (const float*)d_in[1];
  const float* wk   = (const float*)d_in[2];
  const float* wv   = (const float*)d_in[3];
  const float* wo_w = (const float*)d_in[4];
  const float* wo_b = (const float*)d_in[5];
  const float* ep_w = (const float*)d_in[6];
  const float* ep_b = (const float*)d_in[7];
  const float* en_g = (const float*)d_in[8];
  const float* en_b = (const float*)d_in[9];
  const float* ln_g = (const float*)d_in[10];
  const float* ln_b = (const float*)d_in[11];

  float* outp  = (float*)d_out;                 // [Mn, Dn]
  float* attn  = outp + (size_t)Mn * Dn;        // [Bn*Hn, Ln, Ln]

  // workspace carve-out
  char*  ws  = (char*)d_ws;
  size_t off = 0;
  auto carve = [&](size_t bytes) -> char* {
    char* p = ws + off;
    off += (bytes + 255) & ~(size_t)255;
    return p;
  };
  bf16*  xb    = (bf16*)carve((size_t)Mn * Dn * 2);
  bf16*  wqb   = (bf16*)carve((size_t)Dn * Dn * 2);
  bf16*  wkb   = (bf16*)carve((size_t)Dn * Dn * 2);
  bf16*  wvb   = (bf16*)carve((size_t)Dn * Dn * 2);
  bf16*  wepb  = (bf16*)carve((size_t)Dn * Dn * 2);
  bf16*  wob   = (bf16*)carve((size_t)Dn * Dn * 2);
  bf16*  qb    = (bf16*)carve((size_t)Mn * Dn * 2);
  bf16*  kb    = (bf16*)carve((size_t)Mn * Dn * 2);
  bf16*  vb    = (bf16*)carve((size_t)Mn * Dn * 2);
  bf16*  vt    = (bf16*)carve((size_t)Bn * Hn * DKn * Ln * 2);
  float* ef32  = (float*)carve((size_t)Mn * Dn * 4);
  bf16*  efb   = (bf16*)carve((size_t)Mn * Dn * 2);
  float* ebias = (float*)carve((size_t)Bn * Ln * Ln * 4);
  bf16*  ctxb  = (bf16*)carve((size_t)Mn * Dn * 2);
  float* out0  = (float*)carve((size_t)Mn * Dn * 4);

  // 1) convert x + weights to bf16
  {
    const int nx4 = Mn * Dn / 4;   // 1,048,576
    const int nw4 = Dn * Dn / 4;   //   262,144
    k_cvt_bf16<<<dim3((nx4 + 255) / 256), dim3(256), 0, stream>>>(x, xb, nx4);
    k_cvt_bf16<<<dim3((nw4 + 255) / 256), dim3(256), 0, stream>>>(wq, wqb, nw4);
    k_cvt_bf16<<<dim3((nw4 + 255) / 256), dim3(256), 0, stream>>>(wk, wkb, nw4);
    k_cvt_bf16<<<dim3((nw4 + 255) / 256), dim3(256), 0, stream>>>(wv, wvb, nw4);
    k_cvt_bf16<<<dim3((nw4 + 255) / 256), dim3(256), 0, stream>>>(ep_w, wepb, nw4);
    k_cvt_bf16<<<dim3((nw4 + 255) / 256), dim3(256), 0, stream>>>(wo_w, wob, nw4);
  }

  // 2) projections: Q/K/V (bf16 out), energy proj (f32 out + bias)
  {
    dim3 g(Dn / 16 / 4 / 8, Mn / 16, 1);   // (2, 256, 1)
    k_gemm_nt<<<g, 256, 0, stream>>>(xb, 0, Dn, wqb, 0, Dn,
                                     nullptr, qb, 0, Dn, nullptr, Dn);
    k_gemm_nt<<<g, 256, 0, stream>>>(xb, 0, Dn, wkb, 0, Dn,
                                     nullptr, kb, 0, Dn, nullptr, Dn);
    k_gemm_nt<<<g, 256, 0, stream>>>(xb, 0, Dn, wvb, 0, Dn,
                                     nullptr, vb, 0, Dn, nullptr, Dn);
    k_gemm_nt<<<g, 256, 0, stream>>>(xb, 0, Dn, wepb, 0, Dn,
                                     ef32, nullptr, 0, Dn, ep_b, Dn);
  }

  // 3) energy-feature layernorm -> bf16 ; transpose V for the ctx GEMM
  k_ln_bf16<<<dim3(Mn), dim3(256), 0, stream>>>(ef32, en_g, en_b, efb);
  k_vtrans<<<dim3(Ln / 64, Bn * Hn), dim3(256), 0, stream>>>(vb, vt);

  // 4) energy bias: ebias[b] = efb[b] @ efb[b]^T (batched over b)
  {
    dim3 g(Ln / 16 / 4 / 8, Ln / 16, Bn);  // (4, 128, 2)
    k_gemm_nt<<<g, 256, 0, stream>>>(efb, (long)Ln * Dn, Dn,
                                     efb, (long)Ln * Dn, Dn,
                                     ebias, nullptr, (long)Ln * Ln, Ln,
                                     nullptr, Dn);
  }

  // 5) fused scores + softmax + ctx (attn written once, straight to d_out)
  {
    const int ldsBytes = 16 * LDSW * 4;    // 131,328 B of the 320 KB WGP pool
    k_attn_fused<<<dim3(Ln / 16, Bn * Hn), dim3(256), ldsBytes, stream>>>(
        qb, kb, vt, ebias, attn, ctxb);
  }

  // 6) output projection: out0 = ctx @ wo^T + wo_b
  {
    dim3 g(Dn / 16 / 4 / 8, Mn / 16, 1);
    k_gemm_nt<<<g, 256, 0, stream>>>(ctxb, 0, Dn, wob, 0, Dn,
                                     out0, nullptr, 0, Dn, wo_b, Dn);
  }

  // 7) final residual layernorm -> f32 output
  k_ln_residual<<<dim3(Mn), dim3(256), 0, stream>>>(out0, x, ln_g, ln_b, outp);
}
